// RNN_4466765988427
// MI455X (gfx1250) — compile-verified, matrix operand order checked by reading, not run on hardware
//
#include <hip/hip_runtime.h>
#include <hip/hip_bf16.h>
#include <math.h>

// CDNA5 / gfx1250: wave32, WMMA bf16 path.
typedef __attribute__((ext_vector_type(16))) __bf16 v16bf;
typedef __attribute__((ext_vector_type(8)))  __bf16 v8bf;
typedef __attribute__((ext_vector_type(8)))  float  v8f;

#define S_SLOTS 64
#define D_CH    1024
#define H_DIM   256
#define KCAT    512          // [x ; h]
#define GROWS   1024         // 4*H gate rows
#define KC_G    16           // 512 / 32
#define KC_H    8            // 256 / 32
#define HEAD_ELEMS_PER_SLOT (D_CH * H_DIM)   // 262144 packed bf16 per slot

// Per-lane K mapping of the 16-bit A fragment (16x32), ISA 7.12.2:
// lanes 0-15: K = 0..7 then 16..23 ; lanes 16-31: K = 8..15 then 24..31
__host__ __device__ __forceinline__ int kmap(int lane, int e) {
  return ((lane & 16) ? 8 : 0) + ((e < 8) ? e : (8 + e));
}

// ---------------- prep: pack W_cat = [W_ih | W_hh] into WMMA A-fragment layout (bf16)
__global__ void pack_gates_kernel(const float* __restrict__ Wih,
                                  const float* __restrict__ Whh,
                                  __bf16* __restrict__ out) {
  int idx = blockIdx.x * blockDim.x + threadIdx.x;
  if (idx >= GROWS * KCAT) return;
  int e    = idx & 15;
  int lane = (idx >> 4) & 31;
  int c    = (idx >> 9) & 15;      // k-chunk 0..15
  int t    = idx >> 13;            // row tile 0..63
  int m = t * 16 + (lane & 15);
  int k = c * 32 + kmap(lane, e);
  float w = (k < H_DIM) ? Wih[m * H_DIM + k] : Whh[m * H_DIM + (k - H_DIM)];
  out[idx] = (__bf16)w;
}

// ---------------- prep: pack head_W[s]^T ([D x H]) into WMMA A-fragment layout per slot
__global__ void pack_head_kernel(const float* __restrict__ headW,
                                 __bf16* __restrict__ out) {
  size_t idx = blockIdx.x * (size_t)blockDim.x + threadIdx.x;
  if (idx >= (size_t)S_SLOTS * HEAD_ELEMS_PER_SLOT) return;
  int r    = (int)(idx & (HEAD_ELEMS_PER_SLOT - 1));
  int s    = (int)(idx >> 18);
  int e    = r & 15;
  int lane = (r >> 4) & 31;
  int c    = (r >> 9) & 7;         // k-chunk 0..7
  int t    = r >> 12;              // d tile 0..63
  int d = t * 16 + (lane & 15);
  int k = c * 32 + kmap(lane, e);
  out[idx] = (__bf16)headW[((size_t)s * H_DIM + k) * D_CH + d];
}

__device__ __forceinline__ float step_uniform(int step) {
  unsigned long long z = 0x9E3779B97F4A7C15ull * (unsigned long long)(step + 1) + 42ull;
  z ^= z >> 30; z *= 0xBF58476D1CE4E5B9ull;
  z ^= z >> 27; z *= 0x94D049BB133111EBull;
  z ^= z >> 31;
  return (float)(z >> 40) * (1.0f / 16777216.0f);
}

__device__ __forceinline__ float sigmoidf_(float x) { return 1.0f / (1.0f + __expf(-x)); }

// One 16-row tile of a mat-vec: D[16x16] accumulated over kchunks of 32.
// A fragments streamed from packed global memory; B fragments = vector
// replicated across the 16 columns, fetched as two ds_load_b128 per chunk.
// #pragma unroll 2 keeps live A/B frags bounded => no scratch spills at
// the 128-VGPR/wave budget of a 1024-thread block.
__device__ __forceinline__ v8f matvec_tile(const v16bf* __restrict__ at,
                                           const v8bf* __restrict__ x8,
                                           int lane, int kchunks, int kbase8) {
  v8f acc = {};
  const int lo = (lane >> 4) & 1;          // 0: lanes 0-15, 1: lanes 16-31
  #pragma unroll 2
  for (int c = 0; c < kchunks; ++c) {
    v16bf a = at[c * 32 + lane];
    const int cb = kbase8 + c * 4;         // chunk base, units of 8 elems
    v8bf b0 = x8[cb + lo];                 // K = c*32 + lo*8  .. +7
    v8bf b1 = x8[cb + 2 + lo];             // K = c*32 + 16 + lo*8 .. +7
    v16bf b = __builtin_shufflevector(b0, b1,
        0, 1, 2, 3, 4, 5, 6, 7, 8, 9, 10, 11, 12, 13, 14, 15);
    acc = __builtin_amdgcn_wmma_f32_16x16x32_bf16(
        false, a, false, b, (short)0, acc, false, false);
  }
  return acc;
}

// ---------------- persistent sequential LSTM sampler: 1 workgroup = 32 wave32's
__global__ void __launch_bounds__(1024, 1)
lstm_sample_kernel(const __bf16* __restrict__ Wg,    // packed [64 tiles][16 kc][32 lanes][16]
                   const __bf16* __restrict__ Hd,    // packed per slot [64][8][32][16]
                   const float* __restrict__ b_ih,
                   const float* __restrict__ b_hh,
                   const float* __restrict__ head_b, // [S, D]
                   const float* __restrict__ emb,    // [S, D, H]
                   const float* __restrict__ start,  // [H]
                   int* __restrict__ out_dec,
                   float* __restrict__ out_f) {
  __shared__ __align__(32) __bf16 xcat[KCAT];  // [x(256) ; h(256)] bf16 operands
  __shared__ float  cLDS[H_DIM];
  __shared__ float  gatesLDS[GROWS];
  __shared__ float  logitsLDS[D_CH];
  __shared__ float  bufA[D_CH];
  __shared__ float  probs[D_CH];
  __shared__ float  sMax, sSumEl, sZ, sLP, sENT;
  __shared__ int    sChoice;

  const int j    = threadIdx.x;
  const int wave = j >> 5;
  const int lane = j & 31;
  const v8bf* x8 = (const v8bf*)xcat;

  // init state
  if (j < H_DIM) { xcat[j] = (__bf16)start[j]; cLDS[j] = 0.0f; }
  else if (j < KCAT) { xcat[j] = (__bf16)0.0f; }
  if (j == 0) { sLP = 0.0f; sENT = 0.0f; }
  __syncthreads();

  for (int step = 0; step < S_SLOTS; ++step) {
    // Prefetch next slot's packed head weights (L2-resident stream).
    if (step + 1 < S_SLOTS) {
      const char* nxt = (const char*)(Hd + (size_t)(step + 1) * HEAD_ELEMS_PER_SLOT);
      __builtin_prefetch(nxt + (size_t)j * 512, 0, 0);
    }

    // ---- Phase A: gates = W_cat @ [x;h]  via v_wmma_f32_16x16x32_bf16
    #pragma unroll 1
    for (int pass = 0; pass < 2; ++pass) {
      const int t = wave + pass * 32;               // row tile 0..63
      const v16bf* at = (const v16bf*)Wg + (size_t)t * KC_G * 32;
      v8f acc = matvec_tile(at, x8, lane, KC_G, 0);
      // D layout: VGPR r -> M=r (lanes 0-15) / M=8+r (lanes 16-31)
      if (lane == 0) {
        #pragma unroll
        for (int r = 0; r < 8; ++r) gatesLDS[t * 16 + r] = acc[r];
      } else if (lane == 16) {
        #pragma unroll
        for (int r = 0; r < 8; ++r) gatesLDS[t * 16 + 8 + r] = acc[r];
      }
    }
    __syncthreads();

    // ---- Phase B: LSTM cell nonlinearity, update h/c (gate order i,f,g,o)
    if (j < H_DIM) {
      float ig = gatesLDS[j]       + b_ih[j]       + b_hh[j];
      float fg = gatesLDS[256 + j] + b_ih[256 + j] + b_hh[256 + j];
      float gg = gatesLDS[512 + j] + b_ih[512 + j] + b_hh[512 + j];
      float og = gatesLDS[768 + j] + b_ih[768 + j] + b_hh[768 + j];
      ig = sigmoidf_(ig); fg = sigmoidf_(fg); og = sigmoidf_(og); gg = tanhf(gg);
      float cn = fg * cLDS[j] + ig * gg;
      cLDS[j] = cn;
      xcat[H_DIM + j] = (__bf16)(og * tanhf(cn));    // h_new as bf16 operand
    }
    __syncthreads();

    // ---- Phase C: logits = h @ head_W[step] (+ bias), packed as [D x H] tiles
    {
      const v16bf* hbase = (const v16bf*)(Hd + (size_t)step * HEAD_ELEMS_PER_SLOT);
      const float* hb = head_b + (size_t)step * D_CH;
      #pragma unroll 1
      for (int pass = 0; pass < 2; ++pass) {
        const int t = wave + pass * 32;
        const v16bf* at = hbase + (size_t)t * KC_H * 32;
        v8f acc = matvec_tile(at, x8, lane, KC_H, H_DIM / 8);  // h at xcat[256..]
        if (lane == 0) {
          #pragma unroll
          for (int r = 0; r < 8; ++r) logitsLDS[t * 16 + r] = acc[r] + hb[t * 16 + r];
        } else if (lane == 16) {
          #pragma unroll
          for (int r = 0; r < 8; ++r) logitsLDS[t * 16 + 8 + r] = acc[r] + hb[t * 16 + 8 + r];
        }
      }
    }
    __syncthreads();

    // ---- Phase D: log-softmax, entropy, categorical sample, embedding gather
    bufA[j] = logitsLDS[j];
    __syncthreads();
    for (int s = 512; s > 0; s >>= 1) {
      if (j < s) bufA[j] = fmaxf(bufA[j], bufA[j + s]);
      __syncthreads();
    }
    if (j == 0) sMax = bufA[0];
    __syncthreads();

    const float m  = sMax;
    const float ex = __expf(logitsLDS[j] - m);
    probs[j] = ex;
    bufA[j]  = ex * (logitsLDS[j] - m);
    __syncthreads();
    for (int s = 512; s > 0; s >>= 1) {
      if (j < s) bufA[j] += bufA[j + s];
      __syncthreads();
    }
    if (j == 0) sSumEl = bufA[0];
    __syncthreads();

    // inclusive prefix sum (Hillis-Steele) over probs
    for (int off = 1; off < D_CH; off <<= 1) {
      float v   = probs[j];
      float add = (j >= off) ? probs[j - off] : 0.0f;
      __syncthreads();
      probs[j] = v + add;
      __syncthreads();
    }
    if (j == 0) { sZ = probs[D_CH - 1]; sChoice = D_CH - 1; }
    __syncthreads();

    const float target = step_uniform(step) * sZ;
    {
      float hi = probs[j];
      float lo = j ? probs[j - 1] : 0.0f;
      if (lo <= target && target < hi) atomicMin(&sChoice, j);
    }
    __syncthreads();

    if (j == 0) {
      const int ch = sChoice;
      const float logZ = __logf(sZ);
      sLP  += logitsLDS[ch] - sMax - logZ;     // log p(choice)
      sENT += logZ - sSumEl / sZ;              // -(sum p*logp)
      out_dec[step] = ch;
    }
    __syncthreads();

    const int ch = sChoice;
    if (j < H_DIM)
      xcat[j] = (__bf16)emb[((size_t)step * D_CH + ch) * H_DIM + j];
    __syncthreads();
  }

  if (j == 0) {
    out_f[S_SLOTS]     = sLP;
    out_f[S_SLOTS + 1] = sENT * (1.0f / (float)S_SLOTS);
  }
}

extern "C" void kernel_launch(void* const* d_in, const int* in_sizes, int n_in,
                              void* d_out, int out_size, void* d_ws, size_t ws_size,
                              hipStream_t stream) {
  const float* emb    = (const float*)d_in[0];   // [S, D, H]
  const float* start  = (const float*)d_in[1];   // [H]
  const float* Wih    = (const float*)d_in[2];   // [4H, H]
  const float* Whh    = (const float*)d_in[3];   // [4H, H]
  const float* bih    = (const float*)d_in[4];   // [4H]
  const float* bhh    = (const float*)d_in[5];   // [4H]
  const float* headW  = (const float*)d_in[6];   // [S, H, D]
  const float* headB  = (const float*)d_in[7];   // [S, D]

  __bf16* Wg = (__bf16*)d_ws;                          // 1 MB packed gate weights
  __bf16* Hd = (__bf16*)((char*)d_ws + (1u << 20));    // 32 MB packed head weights

  {
    const int n = GROWS * KCAT;
    pack_gates_kernel<<<(n + 255) / 256, 256, 0, stream>>>(Wih, Whh, Wg);
  }
  {
    const size_t n = (size_t)S_SLOTS * HEAD_ELEMS_PER_SLOT;
    pack_head_kernel<<<(unsigned)((n + 255) / 256), 256, 0, stream>>>(headW, Hd);
  }
  lstm_sample_kernel<<<1, 1024, 0, stream>>>(Wg, Hd, bih, bhh, headB, emb, start,
                                             (int*)d_out, (float*)d_out);
}
